// GraphLoss_65481071400836
// MI455X (gfx1250) — matrix operand order repaired and checked by reading, not compiled, not run on hardware
//
#include <hip/hip_runtime.h>
#include <math.h>

// ---- CDNA5 (gfx1250) wave32 WMMA types ----
typedef _Float16 v4h  __attribute__((ext_vector_type(4)));
typedef _Float16 v8h  __attribute__((ext_vector_type(8)));
typedef _Float16 v16h __attribute__((ext_vector_type(16)));
typedef float    v8f  __attribute__((ext_vector_type(8)));

#define D_FEAT 128
#define WAVES_PER_BLOCK 8   // 256 threads = 8 wave32

// ---------------------------------------------------------------------------
// Kernel 1: row-normalize states (f32) and store as f16 rows in workspace.
// One wave per node: 32 lanes x float4 = 128 floats.
// sim = dot(s,d)/((|s|+eps)(|d|+eps)) == dot(normed_s, normed_d), so
// pre-normalizing is exact w.r.t. the reference (up to f16 rounding).
// ---------------------------------------------------------------------------
__global__ void GraphLoss_normalize_f16(const float* __restrict__ states,
                                        _Float16* __restrict__ nf16,
                                        int n_nodes) {
  const int lane = threadIdx.x & 31;
  const int wid  = threadIdx.x >> 5;
  const int node = blockIdx.x * WAVES_PER_BLOCK + wid;
  if (node >= n_nodes) return;

  const float4* row = (const float4*)(states + (size_t)node * D_FEAT);
  float4 v = row[lane];                              // global_load_b128
  float ss = v.x * v.x + v.y * v.y + v.z * v.z + v.w * v.w;
#pragma unroll
  for (int off = 16; off > 0; off >>= 1)             // wave32 butterfly
    ss += __shfl_xor(ss, off, 32);
  const float inv = 1.0f / (sqrtf(ss) + 1e-12f);

  v4h h;
  h[0] = (_Float16)(v.x * inv);
  h[1] = (_Float16)(v.y * inv);
  h[2] = (_Float16)(v.z * inv);
  h[3] = (_Float16)(v.w * inv);
  *(v4h*)(nf16 + (size_t)node * D_FEAT + lane * 4) = h;  // 8B store
}

// ---------------------------------------------------------------------------
// Kernel 2: one wave handles 16 edges. Build A (16 src rows, 16x128 f16) and
// B (16 dst rows as columns, 128x16 f16) fragments per the CDNA5 ISA layouts,
// run 4x v_wmma_f32_16x16x32_f16 (K=128); the 16x16 f32 tile's diagonal is
// the 16 per-edge cosine similarities.
//
// ISA 16-bit A 16x32 layout: lane L (m = L&15, kh = L>>4):
//   v16h elems 0..7  = K = kc + kh*8 + [0..7]
//   v16h elems 8..15 = K = kc + 16 + kh*8 + [0..7]
// ISA 16-bit B 32x16 layout: lane L (n = L&15, kh = L>>4):
//   v16h elems 0..7 = K = kc + kh*8 + [0..7], 8..15 = K = kc + 16 + kh*8 + [0..7] (as A)
// C/D 16x16 f32: lanes 0-15 -> N=lane, M=vgpr(0..7); lanes 16-31 -> N=lane-16,
// M=8+vgpr. Diagonal: lanes 0..7 (vgpr=lane) and lanes 24..31 (vgpr=lane-24).
// ---------------------------------------------------------------------------
__device__ __forceinline__ float pick8(const v8f& c, int i) {
  // cndmask ladder (avoid dynamic-index scratch spill)
  float s0 = (i & 1) ? c[1] : c[0];
  float s1 = (i & 1) ? c[3] : c[2];
  float s2 = (i & 1) ? c[5] : c[4];
  float s3 = (i & 1) ? c[7] : c[6];
  float t0 = (i & 2) ? s1 : s0;
  float t1 = (i & 2) ? s3 : s2;
  return (i & 4) ? t1 : t0;
}

__global__ void GraphLoss_edge_wmma(const _Float16* __restrict__ nf16,
                                    const float* __restrict__ ew,
                                    const int* __restrict__ eidx,
                                    float* __restrict__ partials,
                                    int n_edges, int n_tiles) {
  __shared__ float sdata[WAVES_PER_BLOCK];
  const int lane = threadIdx.x & 31;
  const int wid  = threadIdx.x >> 5;
  const int tile = blockIdx.x * WAVES_PER_BLOCK + wid;

  float contrib = 0.0f;
  if (tile < n_tiles) {                 // wave-uniform branch: EXEC stays full
    const int  le  = lane & 15;         // edge slot within tile (row m / col n)
    const int  kh  = lane >> 4;         // K-half select
    const int  eg  = tile * 16 + le;
    const bool inr = eg < n_edges;
    const int  ec  = inr ? eg : 0;      // clamp; masked by w=0

    const int   srcn = eidx[ec];
    const int   dstn = eidx[n_edges + ec];
    const float w    = inr ? ew[ec] : 0.0f;

    const _Float16* pa = nf16 + (size_t)srcn * D_FEAT;
    const _Float16* pb = nf16 + (size_t)dstn * D_FEAT;

    v8f c = {};
#pragma unroll
    for (int kc = 0; kc < D_FEAT; kc += 32) {
      v8h a0 = *(const v8h*)(pa + kc + kh * 8);        // 16B aligned
      v8h a1 = *(const v8h*)(pa + kc + 16 + kh * 8);
      v8h b0 = *(const v8h*)(pb + kc + kh * 8);
      v8h b1 = *(const v8h*)(pb + kc + 16 + kh * 8);
      v16h A = __builtin_shufflevector(a0, a1, 0,1,2,3,4,5,6,7,8,9,10,11,12,13,14,15);
      v16h B = __builtin_shufflevector(b0, b1, 0,1,2,3,4,5,6,7,8,9,10,11,12,13,14,15);
      // (neg_a, A, neg_b, B, c_mod, C, reuse_a, reuse_b)
      c = __builtin_amdgcn_wmma_f32_16x16x32_f16(false, A, false, B,
                                                 (short)0, c, false, false);
    }

    const bool onDiag = (lane < 8) || (lane >= 24);
    const int  di     = ((lane < 8) ? lane : (lane - 24)) & 7;
    const float sim   = pick8(c, di);
    // for diag lanes, le == edge slot, so this lane's own w is the right one
    contrib = onDiag ? w * (1.0f - sim) : 0.0f;
  }

#pragma unroll
  for (int off = 16; off > 0; off >>= 1)
    contrib += __shfl_xor(contrib, off, 32);
  if (lane == 0) sdata[wid] = contrib;
  __syncthreads();
  if (threadIdx.x == 0) {
    float s = 0.0f;
#pragma unroll
    for (int i = 0; i < WAVES_PER_BLOCK; ++i) s += sdata[i];
    partials[blockIdx.x] = s;
  }
}

// ---------------------------------------------------------------------------
// Kernel 3: reduce per-block partials, scale by 1/n_edges, write scalar out.
// (Single block; avoids 12.5k same-address f32 atomics.)
// ---------------------------------------------------------------------------
__global__ void GraphLoss_reduce(const float* __restrict__ partials,
                                 float* __restrict__ out,
                                 int n, float inv_edges) {
  __shared__ float sdata[WAVES_PER_BLOCK];
  float s = 0.0f;
  for (int i = threadIdx.x; i < n; i += blockDim.x) s += partials[i];
#pragma unroll
  for (int off = 16; off > 0; off >>= 1) s += __shfl_xor(s, off, 32);
  const int lane = threadIdx.x & 31, wid = threadIdx.x >> 5;
  if (lane == 0) sdata[wid] = s;
  __syncthreads();
  if (threadIdx.x == 0) {
    float t = 0.0f;
#pragma unroll
    for (int i = 0; i < WAVES_PER_BLOCK; ++i) t += sdata[i];
    out[0] = t * inv_edges;
  }
}

// ---------------------------------------------------------------------------
extern "C" void kernel_launch(void* const* d_in, const int* in_sizes, int n_in,
                              void* d_out, int out_size, void* d_ws, size_t ws_size,
                              hipStream_t stream) {
  const float* states = (const float*)d_in[0];   // [N,128] f32
  const float* ew     = (const float*)d_in[1];   // [E] f32
  const int*   eidx   = (const int*)d_in[2];     // [2,E] int (harness int cast)

  const int n_nodes = in_sizes[0] / D_FEAT;
  const int n_edges = in_sizes[1];

  // Workspace: normalized f16 rows (25.6 MB for N=100k) + per-block partials.
  _Float16* nf16 = (_Float16*)d_ws;
  size_t nf16_bytes = ((size_t)n_nodes * D_FEAT * sizeof(_Float16) + 255) & ~(size_t)255;
  float* partials = (float*)((char*)d_ws + nf16_bytes);

  const int blocks1 = (n_nodes + WAVES_PER_BLOCK - 1) / WAVES_PER_BLOCK;
  GraphLoss_normalize_f16<<<blocks1, 32 * WAVES_PER_BLOCK, 0, stream>>>(
      states, nf16, n_nodes);

  const int n_tiles = (n_edges + 15) / 16;
  const int blocks2 = (n_tiles + WAVES_PER_BLOCK - 1) / WAVES_PER_BLOCK;
  GraphLoss_edge_wmma<<<blocks2, 32 * WAVES_PER_BLOCK, 0, stream>>>(
      nf16, ew, eidx, partials, n_edges, n_tiles);

  GraphLoss_reduce<<<1, 32 * WAVES_PER_BLOCK, 0, stream>>>(
      partials, (float*)d_out, blocks2, 1.0f / (float)n_edges);
}
